// ConvBnAct_670014898567
// MI455X (gfx1250) — matrix operand order repaired and checked by reading, not compiled, not run on hardware
//
#include <hip/hip_runtime.h>
#include <math.h>

// -------- Problem constants (from reference) --------
#define Bsz   32
#define CIN   256
#define COUT  256
#define Hd    56
#define Wd    56
#define HW    (Hd * Wd)          // 3136
#define NEXP  4
#define KTOT  (CIN * 9)          // 2304 = 72*32
#define WPS   (COUT * KTOT)      // 589824 weights per sample
#define BN_EPS 1e-5f
#define NSTEP 72                 // K-steps of 32

// -------- vector types --------
typedef __attribute__((ext_vector_type(16))) __bf16 v16bf;
typedef __attribute__((ext_vector_type(8)))  __bf16 v8bf;
typedef __attribute__((ext_vector_type(8)))  float  v8f;
typedef __attribute__((ext_vector_type(4)))  unsigned int u32x4;
typedef __attribute__((ext_vector_type(8)))  int  i32x8;
typedef __attribute__((ext_vector_type(4)))  int  i32x4;

__device__ __forceinline__ unsigned int f2bf_bits(float f) {
    unsigned int u = __builtin_bit_cast(unsigned int, f);
    u += 0x7FFFu + ((u >> 16) & 1u);
    return u >> 16;
}
// two fp32 -> packed bf16x2 via v_perm_b32 (round-half-up)
__device__ __forceinline__ unsigned int pack_bf16(float a, float b) {
    unsigned int ua = __builtin_bit_cast(unsigned int, a) + 0x8000u;
    unsigned int ub = __builtin_bit_cast(unsigned int, b) + 0x8000u;
    return __builtin_amdgcn_perm(ub, ua, 0x07060302u);
}
__device__ __forceinline__ v16bf frag16(const unsigned short* p0,
                                        const unsigned short* p1) {
    v8bf lo = *(const v8bf*)p0;
    v8bf hi = *(const v8bf*)p1;
    return __builtin_shufflevector(lo, hi,
            0,1,2,3,4,5,6,7,8,9,10,11,12,13,14,15);
}

// ============================================================
// Kernel 1: routing = sigmoid(GAP(x) @ route_w^T + route_b)  [B, E]
// ============================================================
__global__ __launch_bounds__(256) void routing_kernel(
    const float* __restrict__ x, const float* __restrict__ rw,
    const float* __restrict__ rb, float* __restrict__ routing) {
    __shared__ float pooled[CIN];
    __shared__ float red[256];
    const int b    = blockIdx.x;
    const int tid  = threadIdx.x;
    const int wave = tid >> 5;
    const int lane = tid & 31;

    for (int j = 0; j < 32; ++j) {
        const int c = wave * 32 + j;
        const float* p = x + ((size_t)b * CIN + c) * HW;
        float s = 0.f;
        for (int i = lane; i < HW; i += 32) s += p[i];
        for (int m = 16; m > 0; m >>= 1) s += __shfl_xor(s, m, 32);
        if (lane == 0) pooled[c] = s * (1.0f / (float)HW);
    }
    __syncthreads();

    for (int e = 0; e < NEXP; ++e) {
        red[tid] = pooled[tid] * rw[e * CIN + tid];
        __syncthreads();
        for (int s = 128; s > 0; s >>= 1) {
            if (tid < s) red[tid] += red[tid + s];
            __syncthreads();
        }
        if (tid == 0) {
            float z = red[0] + rb[e];
            routing[b * NEXP + e] = 1.0f / (1.0f + __expf(-z));
        }
        __syncthreads();
    }
}

// ============================================================
// Kernel 2: per-sample mixed bf16 weights, K permuted to r-major:
//   wsw[b][co][r*256 + ci] = sum_e routing[b,e] * ew[e][co][ci][r]
// ============================================================
__global__ __launch_bounds__(256) void mix_kernel(
    const float* __restrict__ ew, const float* __restrict__ routing,
    unsigned short* __restrict__ wsw) {
    const unsigned idx = blockIdx.x * 256u + threadIdx.x;   // b*65536 + co*256 + ci
    const int b  = idx >> 16;
    const int co = (idx >> 8) & 255;
    const int ci = idx & 255;

    const float4 r = *(const float4*)(routing + b * NEXP);
    const size_t src = ((size_t)co * CIN + ci) * 9;
    const float* e0 = ew + src;
    const float* e1 = e0 + (size_t)WPS;
    const float* e2 = e1 + (size_t)WPS;
    const float* e3 = e2 + (size_t)WPS;

    unsigned short* dst = wsw + (size_t)b * WPS + (size_t)co * KTOT + ci;
    #pragma unroll
    for (int rr = 0; rr < 9; ++rr) {
        float s = r.x * e0[rr] + r.y * e1[rr] + r.z * e2[rr] + r.w * e3[rr];
        dst[rr * 256] = (unsigned short)f2bf_bits(s);
    }
}

// ============================================================
// Kernel 3: implicit-GEMM 3x3 conv + BN + ReLU + residual
// Block tile M=128 x N=128, K-step 32 (r-major K), 8 waves (4x2),
// wave tile 32x64 -> 8x v_wmma_f32_16x16x32_bf16 per K-step.
// Double-buffered LDS, software pipeline:
//   top:    global loads (regs) + TDM issue for step k -> buf[k&1]
//   middle: 8 WMMAs for step k-1 from buf[(k&1)^1]
//   bottom: pack + ds_store, s_wait_tensorcnt, one barrier
// ============================================================
#define LDSS 40   // 32 data + 8 pad bf16 (TDM pad: 16DW interval, 4DW amount)
#define ABUF (128 * LDSS)                  // shorts per buffer
__global__ __launch_bounds__(256) void conv_kernel(
    const unsigned short* __restrict__ wsw,   // [B][COUT][KTOT] bf16, r-major K
    const float* __restrict__ x,              // [B][CIN][H][W]
    const float* __restrict__ gamma, const float* __restrict__ beta,
    const float* __restrict__ mean,  const float* __restrict__ var,
    float* __restrict__ out) {
    __shared__ __align__(16) unsigned short As[2][128][LDSS];
    __shared__ __align__(16) unsigned short Bs[2][128][LDSS];

    const int tid  = threadIdx.x;
    const int n0   = blockIdx.x * 128;     // 25 tiles, last one partial
    const int co0  = blockIdx.y * 128;     // 2 tiles, exact
    const int b    = blockIdx.z;

    const int wave = tid >> 5;
    const int lane = tid & 31;
    const int wm   = wave & 3;             // 32-row M subtile
    const int wn   = wave >> 2;            // 64-col N subtile
    const int lh   = lane >> 4;
    const int l    = lane & 15;

    // B-tile fill: 256 threads cover 128(n) x 32(k), 16 consecutive k per thread
    const int fN = tid & 127;
    const int ks = (tid >> 7) * 16;

    const size_t xbase = (size_t)b * CIN * HW;
    const int n  = n0 + fN;
    const int oh = n / Wd;
    const int ow = n - oh * Wd;
    const bool nvalid = n < HW;

    // ---- TDM descriptor (loop-invariant parts) ----
    const unsigned lds_as = (unsigned)(uintptr_t)(&As[0][0][0]);
    const unsigned long long wbytes =
        (unsigned long long)(uintptr_t)wsw +
        2ull * ((size_t)b * WPS + (size_t)co0 * KTOT);
    i32x8 g1;
    g1[0] = (int)((1u << 16) | (1u << 20) | (3u << 22) | (3u << 25));
    g1[1] = (int)(32u << 16);     // tensor_dim0 = 32
    g1[2] = (int)(128u << 16);    // tensor_dim1 = 128
    g1[3] = (int)(32u << 16);     // tile_dim0 = 32
    g1[4] = (int)128u;            // tile_dim1 = 128
    g1[5] = (int)2304u;           // tensor_dim0_stride
    g1[6] = 0;
    g1[7] = 0;
    const i32x4 gz4 = {0, 0, 0, 0};
    const i32x8 gz8 = {0, 0, 0, 0, 0, 0, 0, 0};

    v8f acc[2][4];
    #pragma unroll
    for (int i = 0; i < 2; ++i)
        #pragma unroll
        for (int j = 0; j < 4; ++j) acc[i][j] = (v8f){};

    for (int kidx = 0; kidx <= NSTEP; ++kidx) {
        const bool doStage   = kidx < NSTEP;
        const bool doCompute = kidx > 0;
        const int  q  = kidx & 1;          // buffer being staged
        const int  qc = q ^ 1;             // buffer being computed

        // ---------- stage step kidx: global loads + TDM issue ----------
        float vr[16];
        if (doStage) {
            const int r  = kidx >> 3;            // 0..8 kernel position
            const int ct = (kidx & 7) << 5;      // channel block
            const int kh = (r * 11) >> 5;        // r/3 for r<9
            const int kw = r - kh * 3;
            const int ih = oh + kh - 1;
            const int iw = ow + kw - 1;
            const bool ok = nvalid &
                ((unsigned)ih < (unsigned)Hd) & ((unsigned)iw < (unsigned)Wd);
            #pragma unroll
            for (int i = 0; i < 16; ++i) vr[i] = 0.f;
            if (ok) {
                const float* p = x + xbase + (ih * Wd + iw) + (size_t)(ct + ks) * HW;
                #pragma unroll
                for (int i = 0; i < 16; ++i) vr[i] = p[(size_t)i * HW];
            }
            if (wave == 0) {
                const unsigned long long ga = wbytes + 64ull * (unsigned)kidx;
                u32x4 g0;
                g0[0] = 1u;
                g0[1] = lds_as + (unsigned)(q * ABUF * 2);
                g0[2] = (unsigned)ga;
                g0[3] = (unsigned)(ga >> 32) | (2u << 30);
                __builtin_amdgcn_tensor_load_to_lds(g0, g1, gz4, gz4, gz8, 0);
            }
        }

        // ---------- compute step kidx-1 from buf[qc] ----------
        if (doCompute) {
            const unsigned short (*Aq)[LDSS] = As[qc];
            const unsigned short (*Bq)[LDSS] = Bs[qc];
            const int ak = lh * 8;
            const int bk = lh * 8;
            v16bf a0 = frag16(&Aq[wm * 32 + l][ak],      &Aq[wm * 32 + l][ak + 16]);
            v16bf a1 = frag16(&Aq[wm * 32 + 16 + l][ak], &Aq[wm * 32 + 16 + l][ak + 16]);
            #pragma unroll
            for (int j = 0; j < 4; ++j) {
                const int col = wn * 64 + j * 16 + l;
                v16bf bv = frag16(&Bq[col][bk], &Bq[col][bk + 16]);
                acc[0][j] = __builtin_amdgcn_wmma_f32_16x16x32_bf16(
                                false, a0, false, bv, (short)0, acc[0][j], false, false);
                acc[1][j] = __builtin_amdgcn_wmma_f32_16x16x32_bf16(
                                false, a1, false, bv, (short)0, acc[1][j], false, false);
            }
        }

        // ---------- finish stage: pack + LDS store, tensor wait ----------
        if (doStage) {
            uint4 p0, p1;
            p0.x = pack_bf16(vr[0],  vr[1]);  p0.y = pack_bf16(vr[2],  vr[3]);
            p0.z = pack_bf16(vr[4],  vr[5]);  p0.w = pack_bf16(vr[6],  vr[7]);
            p1.x = pack_bf16(vr[8],  vr[9]);  p1.y = pack_bf16(vr[10], vr[11]);
            p1.z = pack_bf16(vr[12], vr[13]); p1.w = pack_bf16(vr[14], vr[15]);
            *(uint4*)&Bs[q][fN][ks]     = p0;
            *(uint4*)&Bs[q][fN][ks + 8] = p1;
            if (wave == 0) __builtin_amdgcn_s_wait_tensorcnt(0);
        }
        __syncthreads();
    }

    // ---- epilogue: BN(eval) + ReLU + residual (mask partial N tile) ----
    #pragma unroll
    for (int i = 0; i < 2; ++i) {
        #pragma unroll
        for (int j = 0; j < 8; ++j) {
            const int co = co0 + wm * 32 + i * 16 + lh * 8 + j;
            const float inv  = gamma[co] * rsqrtf(var[co] + BN_EPS);
            const float bias = beta[co] - mean[co] * inv;
            const size_t obase = ((size_t)b * COUT + co) * HW;
            #pragma unroll
            for (int jt = 0; jt < 4; ++jt) {
                const int ncol = n0 + wn * 64 + jt * 16 + l;
                if (ncol < HW) {
                    const size_t o = obase + ncol;
                    out[o] = fmaxf(acc[i][jt][j] * inv + bias, 0.f) + x[o];
                }
            }
        }
    }
}

// ============================================================
// Launch. ws: [0,512): routing fp32; [1024, +37.75MB): bf16 mixed weights
// ============================================================
extern "C" void kernel_launch(void* const* d_in, const int* in_sizes, int n_in,
                              void* d_out, int out_size, void* d_ws, size_t ws_size,
                              hipStream_t stream) {
    const float* x   = (const float*)d_in[0];
    const float* rw  = (const float*)d_in[1];
    const float* rb  = (const float*)d_in[2];
    const float* ew  = (const float*)d_in[3];
    const float* g   = (const float*)d_in[4];
    const float* be  = (const float*)d_in[5];
    const float* mu  = (const float*)d_in[6];
    const float* va  = (const float*)d_in[7];
    float* out = (float*)d_out;

    float* routing      = (float*)d_ws;
    unsigned short* wsw = (unsigned short*)((char*)d_ws + 1024);

    routing_kernel<<<Bsz, 256, 0, stream>>>(x, rw, rb, routing);

    const int mixBlocks = (Bsz * COUT * CIN) / 256;   // 8192
    mix_kernel<<<mixBlocks, 256, 0, stream>>>(ew, routing, wsw);

    dim3 grid((HW + 127) / 128, COUT / 128, Bsz);     // 25 x 2 x 32
    conv_kernel<<<grid, 256, 0, stream>>>(wsw, x, g, be, mu, va, out);
}
